// FakeLSTM_25709674234179
// MI455X (gfx1250) — matrix operand order repaired and checked by reading, not compiled, not run on hardware
//
#include <hip/hip_runtime.h>
#include <hip/hip_bf16.h>

#define INPUT_SZ 2048
#define HID      2048
#define ITERS    256
#define NROW     (4 * HID)          // 8192 gate rows
#define KDIM     (INPUT_SZ + HID)   // 4096 combined K

typedef __attribute__((ext_vector_type(16))) __bf16 v16bf;
typedef __attribute__((ext_vector_type(8)))  __bf16 v8bf;
typedef __attribute__((ext_vector_type(8)))  float  v8f;

// ---- one-time (per launch) weight conversion: Wc[row][0:2048)=Wi, [2048:4096)=Wh, bf16 ----
__global__ void convert_w_kernel(const float* __restrict__ Wi,
                                 const float* __restrict__ Wh,
                                 __bf16* __restrict__ Wc) {
  const size_t total = (size_t)NROW * KDIM;
  for (size_t idx = blockIdx.x * (size_t)blockDim.x + threadIdx.x; idx < total;
       idx += (size_t)gridDim.x * blockDim.x) {
    size_t row = idx >> 12;       // /4096
    size_t col = idx & 4095;
    float v = (col < INPUT_SZ) ? Wi[row * INPUT_SZ + col]
                               : Wh[row * HID + (col - INPUT_SZ)];
    Wc[idx] = (__bf16)v;
  }
}

// ---- transpose+convert x: xb[t*2048 + i] = bf16(x[i*256 + t]) ----
__global__ void convert_x_kernel(const float* __restrict__ x, __bf16* __restrict__ xb) {
  const size_t total = (size_t)INPUT_SZ * ITERS;
  for (size_t idx = blockIdx.x * (size_t)blockDim.x + threadIdx.x; idx < total;
       idx += (size_t)gridDim.x * blockDim.x) {
    size_t t = idx >> 11;         // /2048
    size_t i = idx & 2047;
    xb[idx] = (__bf16)x[i * ITERS + t];
  }
}

// ---- zero recurrent state + h_states column 0 ----
__global__ void init_state_kernel(float* __restrict__ A_f32, float* __restrict__ B_f32,
                                  __bf16* __restrict__ Abf0, float* __restrict__ hst) {
  int i = blockIdx.x * blockDim.x + threadIdx.x;
  if (i < HID) {
    A_f32[i] = 0.f;
    B_f32[i] = 0.f;
    Abf0[i] = (__bf16)0.f;
    hst[(size_t)i * (ITERS + 1)] = 0.f;   // h_states[:,0] = 0
  }
}

// ---- one LSTM timestep: 128 blocks x 512 threads (16 waves) ----
// wave w: gate g = w>>2, K-quarter q = w&3 (1024 K each, 32 WMMA of K=32).
// GEMV-as-GEMM: only column 0 of the B tile / D tile is meaningful; all other
// columns carry don't-care data (columns of D are independent, never read).
__global__ __launch_bounds__(512) void lstm_step_kernel(
    const __bf16* __restrict__ Wc,        // 8192 x 4096 bf16, row-major
    const __bf16* __restrict__ xb_t,      // x_t bf16 [2048]
    const __bf16* __restrict__ Abf_prev,  // recurrent vec bf16 [2048]
    __bf16* __restrict__ Abf_next,        // next recurrent vec bf16 [2048]
    const float* __restrict__ bi, const float* __restrict__ bh,
    float* __restrict__ A_f32, float* __restrict__ B_f32,
    float* __restrict__ hcol)             // &h_states[0][t+1]; stride 257
{
  __shared__ float partial[16][16];

  const int tid  = threadIdx.x;
  const int wave = tid >> 5;
  const int lane = tid & 31;
  const int g = wave >> 2;                 // gate 0..3
  const int q = wave & 3;                  // K quarter 0..3
  const int h0 = blockIdx.x * 16;
  const int rowbase = g * HID + h0;
  const int r  = lane & 15;                // tile row owned by this lane
  const int kb = (lane >> 4) << 4;         // 0 or 16 (contiguous K sub-offset)

  // weight row pointer for this lane, offset to this wave's K-quarter (+ kb)
  const __bf16* ap = Wc + (size_t)(rowbase + r) * KDIM + q * 1024 + kb;
  // vector source for this K-quarter: quarters 0,1 -> x_t; 2,3 -> A_prev.
  // Lanes 0-15 hold B-tile K=0..15, lanes 16-31 hold K=16..31; all lanes of a
  // half load the same (valid) address — non-column-0 results are discarded.
  const __bf16* vp = ((q < 2) ? (xb_t + q * 1024) : (Abf_prev + (q - 2) * 1024))
                     + ((lane >> 4) << 4);

  v8f c = {};
#pragma unroll
  for (int it = 0; it < 32; ++it) {
    const int k0 = it * 32;
    // A tile (16x32 bf16): lane holds row r, contiguous 16 elems at kb
    v16bf a = *(const v16bf*)(ap + k0);
    v16bf b = *(const v16bf*)(vp + k0);
    c = __builtin_amdgcn_wmma_f32_16x16x32_bf16(
            /*neg_a=*/false, a, /*neg_b=*/false, b,
            /*c_mod=*/(short)0, c, /*reuse_a=*/false, /*reuse_b=*/false);
  }

  // column 0 of C/D: lane 0 holds M=0..7 in c[0..7], lane 16 holds M=8..15
  if (lane == 0) {
#pragma unroll
    for (int m = 0; m < 8; ++m) partial[wave][m] = c[m];
  } else if (lane == 16) {
#pragma unroll
    for (int m = 0; m < 8; ++m) partial[wave][8 + m] = c[m];
  }
  __syncthreads();

  // pointwise gate update for the block's 16 hidden units
  if (tid < 16) {
    const int j = h0 + tid;
    float gate[4];
#pragma unroll
    for (int gg = 0; gg < 4; ++gg) {
      float s = partial[4 * gg + 0][tid] + partial[4 * gg + 1][tid] +
                partial[4 * gg + 2][tid] + partial[4 * gg + 3][tid];
      const int row = gg * HID + j;
      gate[gg] = s + bi[row] + bh[row];
    }
    const float gi = 1.f / (1.f + __expf(-gate[0]));
    const float gf = 1.f / (1.f + __expf(-gate[1]));
    const float ct = tanhf(gate[2]);
    const float go = 1.f / (1.f + __expf(-gate[3]));

    const float Bprev = B_f32[j];
    const float Anew = gf * Bprev + gi * ct;   // state_cell (feeds next matmul)
    const float Bnew = go * tanhf(ct);         // state_hidden (emitted)

    A_f32[j] = Anew;
    Abf_next[j] = (__bf16)Anew;
    B_f32[j] = Bnew;
    hcol[(size_t)j * (ITERS + 1)] = Bnew;      // h_states[j][t+1]
  }
}

// ---- final linear: out[j] = lin_b[j] + sum_k B_final[k] * lin_w[j][k] ----
__global__ void final_linear_kernel(const float* __restrict__ lin_w,
                                    const float* __restrict__ lin_b,
                                    const float* __restrict__ Bst,
                                    float* __restrict__ out) {
  const int gwave = (int)((blockIdx.x * (size_t)blockDim.x + threadIdx.x) >> 5);
  const int lane = threadIdx.x & 31;
  if (gwave >= INPUT_SZ) return;
  const float* wrow = lin_w + (size_t)gwave * HID;
  float s = 0.f;
  for (int k = lane; k < HID; k += 32) s += wrow[k] * Bst[k];
#pragma unroll
  for (int off = 16; off > 0; off >>= 1) s += __shfl_down(s, off, 32);
  if (lane == 0) out[gwave] = s + lin_b[gwave];
}

// ---- copy final carries: hs_prev := A_final, cs_prev := B_final ----
__global__ void copy_states_kernel(const float* __restrict__ A_f32,
                                   const float* __restrict__ B_f32,
                                   float* __restrict__ hs_out,
                                   float* __restrict__ cs_out) {
  int i = blockIdx.x * blockDim.x + threadIdx.x;
  if (i < HID) {
    hs_out[i] = A_f32[i];
    cs_out[i] = B_f32[i];
  }
}

extern "C" void kernel_launch(void* const* d_in, const int* in_sizes, int n_in,
                              void* d_out, int out_size, void* d_ws, size_t ws_size,
                              hipStream_t stream) {
  const float* x   = (const float*)d_in[0];   // (2048, 256)
  const float* Wi  = (const float*)d_in[1];   // (8192, 2048)
  const float* Wh  = (const float*)d_in[2];   // (8192, 2048)
  const float* bi  = (const float*)d_in[3];   // (8192, 1)
  const float* bh  = (const float*)d_in[4];   // (8192, 1)
  const float* lw  = (const float*)d_in[5];   // (2048, 2048)
  const float* lb  = (const float*)d_in[6];   // (2048,)
  float* out = (float*)d_out;

  // workspace layout
  char* ws = (char*)d_ws;
  __bf16* Wc  = (__bf16*)ws;                                   // 64 MB
  __bf16* xb  = (__bf16*)(ws + (size_t)NROW * KDIM * 2);       // 1 MB
  __bf16* Abf = (__bf16*)((char*)xb + (size_t)INPUT_SZ * ITERS * 2);  // 2 x 2048 bf16
  float* A_f32 = (float*)((char*)Abf + 2 * HID * sizeof(__bf16));
  float* B_f32 = A_f32 + HID;

  // output layout (flat concat): output[2048] | h_states[2048*257] | hs_prev[2048] | cs_prev[2048]
  float* hst = out + 2048;                                     // h_states, row-major (2048, 257)
  float* hs_out = out + 2048 + (size_t)HID * (ITERS + 1);
  float* cs_out = hs_out + HID;

  convert_w_kernel<<<4096, 256, 0, stream>>>(Wi, Wh, Wc);
  convert_x_kernel<<<512, 256, 0, stream>>>(x, xb);
  init_state_kernel<<<(HID + 255) / 256, 256, 0, stream>>>(A_f32, B_f32, Abf, hst);

  int cur = 0;
  for (int t = 0; t < ITERS; ++t) {
    lstm_step_kernel<<<HID / 16, 512, 0, stream>>>(
        Wc, xb + (size_t)t * INPUT_SZ,
        Abf + (size_t)cur * HID, Abf + (size_t)(cur ^ 1) * HID,
        bi, bh, A_f32, B_f32, hst + (t + 1));
    cur ^= 1;
  }

  final_linear_kernel<<<(INPUT_SZ * 32) / 256, 256, 0, stream>>>(lw, lb, B_f32, out);
  copy_states_kernel<<<(HID + 255) / 256, 256, 0, stream>>>(A_f32, B_f32, hs_out, cs_out);
}
